// Attention_73787538145300
// MI455X (gfx1250) — compile-verified, matrix operand order checked by reading, not run on hardware
//
#include <hip/hip_runtime.h>
#include <hip/hip_bf16.h>
#include <stdint.h>

#define DIM   768
#define HEADS 12
#define HD    64
#define SEQ   2048
#define NB    4
#define QK_SCALE 0.125f   // 64^-0.5

typedef _Float16 v16h __attribute__((ext_vector_type(16)));
typedef _Float16 h8   __attribute__((ext_vector_type(8)));
typedef float    v8f  __attribute__((ext_vector_type(8)));
typedef uint32_t u32x4 __attribute__((ext_vector_type(4)));
typedef uint32_t u32x8 __attribute__((ext_vector_type(8)));

// CDNA5 16-bit A/B fragment layout (wave32):
//  lanes 0-15 : K = {0..7, 16..23};  lanes 16-31 : K = {8..15, 24..31}
//  -> per lane: two contiguous 8-half (16B) chunks at +hi*8 and +16+hi*8.
__device__ __forceinline__ v16h load_frag_f16(const _Float16* p, int hi) {
  h8 c0 = *reinterpret_cast<const h8*>(p + hi * 8);
  h8 c1 = *reinterpret_cast<const h8*>(p + 16 + hi * 8);
  v16h r;
#pragma unroll
  for (int i = 0; i < 8; ++i) { r[i] = c0[i]; r[8 + i] = c1[i]; }
  return r;
}

#define WMMA_F16(a, b, c) \
  __builtin_amdgcn_wmma_f32_16x16x32_f16(false, (a), false, (b), (short)0, (c), false, false)

// ---------------------------------------------------------------------------
// TDM descriptor builders (CDNA5 ISA §8.3/8.4, 2D tensors, group2/3 = NULL)
// ---------------------------------------------------------------------------
__device__ __forceinline__ u32x4 tdm_group0(uint32_t lds_off, uint64_t gaddr) {
  u32x4 g;
  g[0] = 1u;                                      // count=1, user mode
  g[1] = lds_off;                                 // lds_addr (bytes)
  g[2] = (uint32_t)gaddr;                         // global_addr[31:0]
  g[3] = (uint32_t)(gaddr >> 32) | (2u << 30);    // global_addr[56:32] | type=2
  return g;
}

__device__ __forceinline__ u32x8 tdm_group1(uint32_t tensor_d0, uint32_t tensor_d1,
                                            uint32_t tile_d0, uint32_t tile_d1,
                                            uint32_t stride_d0) {
  u32x8 g;
  g[0] = (1u << 16);                              // data_size = 1 -> 2 bytes
  g[1] = (tensor_d0 & 0xFFFFu) << 16;             // tensor_dim0 [79:48]
  g[2] = (tensor_d0 >> 16) | ((tensor_d1 & 0xFFFFu) << 16);   // tensor_dim1 [111:80]
  g[3] = (tensor_d1 >> 16) | ((tile_d0 & 0xFFFFu) << 16);     // tile_dim0 [127:112]
  g[4] = (tile_d1 & 0xFFFFu);                     // tile_dim1 [143:128], tile_dim2=0
  g[5] = stride_d0;                               // tensor_dim0_stride [191:160]
  g[6] = 0;
  g[7] = 0;
  return g;
}

// ---------------------------------------------------------------------------
// Pack kernels (run once): fp32 -> f16, weights transposed so B-fragments
// are two contiguous b128 loads per lane.
// ---------------------------------------------------------------------------
__global__ __launch_bounds__(256) void cvt_f32_f16_kernel(
    const float* __restrict__ src, _Float16* __restrict__ dst, int n) {
  int i = blockIdx.x * 256 + threadIdx.x;
  if (i < n) dst[i] = (_Float16)src[i];
}

__global__ __launch_bounds__(256) void transpose_cvt_kernel(
    const float* __restrict__ W, _Float16* __restrict__ Wt) {
  int i = blockIdx.x * 256 + threadIdx.x;      // coalesced read along n
  int k = i / DIM, n = i % DIM;
  Wt[(size_t)n * DIM + k] = (_Float16)W[i];
}

// ---------------------------------------------------------------------------
// Kernel 1: fused QKV projection (f16 A, f16 transposed B).
// Each wave computes a 16x32 tile (A fragment reused for 2 WMMAs).
//  Q -> f16 [bh][s][64]  (pre-scaled)   K -> f16 [bh][s][64]
//  V -> f16 [bh][64][s]  (transposed so PV B-fragments are contiguous)
// ---------------------------------------------------------------------------
__global__ __launch_bounds__(256) void qkv_proj_kernel(
    const _Float16* __restrict__ Xf, const _Float16* __restrict__ Xt,
    const _Float16* __restrict__ Wtq, const float* __restrict__ bq,
    const _Float16* __restrict__ Wtk, const float* __restrict__ bk,
    const _Float16* __restrict__ Wtv, const float* __restrict__ bv,
    _Float16* __restrict__ Qh, _Float16* __restrict__ Kh,
    _Float16* __restrict__ Vt)
{
  const int which = blockIdx.z;
  const int b     = blockIdx.y;
  const int wave  = threadIdx.x >> 5;
  const int lane  = threadIdx.x & 31;
  const int hi    = lane >> 4;
  const int ln    = lane & 15;

  const int t  = blockIdx.x * 8 + wave;          // 3072 tiles per (b,which)
  const int tm = (t / (DIM / 32)) * 16;
  const int tn = (t % (DIM / 32)) * 32;

  const _Float16* X    = (which == 0) ? Xf : Xt;
  const _Float16* Wt   = (which == 0) ? Wtq : (which == 1) ? Wtk : Wtv;
  const float*    bias = (which == 0) ? bq : (which == 1) ? bk : bv;

  const _Float16* arow = X + ((size_t)b * SEQ + tm + ln) * DIM;
  const _Float16* w0p  = Wt + (size_t)(tn + ln) * DIM;
  const _Float16* w1p  = Wt + (size_t)(tn + 16 + ln) * DIM;

  v8f acc0 = {}, acc1 = {};
#pragma unroll 4
  for (int k0 = 0; k0 < DIM; k0 += 32) {
    v16h a  = load_frag_f16(arow + k0, hi);
    v16h b0 = load_frag_f16(w0p + k0, hi);
    v16h b1 = load_frag_f16(w1p + k0, hi);
    acc0 = WMMA_F16(a, b0, acc0);
    acc1 = WMMA_F16(a, b1, acc1);
  }

#pragma unroll
  for (int half = 0; half < 2; ++half) {
    const int   col = tn + half * 16 + ln;
    const float bb  = bias[col];
    const int   h   = col >> 6;
    const int   d   = col & 63;
    const int   bh  = b * HEADS + h;
    const v8f&  acc = half ? acc1 : acc0;
#pragma unroll
    for (int r = 0; r < 8; ++r) {
      const int   row = tm + r + hi * 8;
      const float v   = acc[r] + bb;
      if (which == 0)
        Qh[((size_t)bh * SEQ + row) * HD + d] = (_Float16)(v * QK_SCALE);
      else if (which == 1)
        Kh[((size_t)bh * SEQ + row) * HD + d] = (_Float16)v;
      else
        Vt[((size_t)bh * HD + d) * SEQ + row] = (_Float16)v;
    }
  }
}

// ---------------------------------------------------------------------------
// Kernel 2: flash attention.  grid=(Sf/64, H, B), block=128 (4 waves).
// Wave 0 stages the shared K (32x64) and V (64x32) tiles into LDS with the
// Tensor Data Mover; all 4 waves consume them via ds_load_b128 fragments.
// ---------------------------------------------------------------------------
__global__ __launch_bounds__(128) void flash_attn_kernel(
    const _Float16* __restrict__ Qh, const _Float16* __restrict__ Kh,
    const _Float16* __restrict__ Vt, const int* __restrict__ mask,
    _Float16* __restrict__ O16)
{
  __shared__ __align__(16) _Float16 ldsK[32 * HD];     // 32 keys x 64 dims
  __shared__ __align__(16) _Float16 ldsV[HD * 32];     // 64 dims x 32 keys
  __shared__ __align__(16) _Float16 ldsP[4][16 * 32];  // wave-private P tiles

  const int b    = blockIdx.z;
  const int h    = blockIdx.y;
  const int wave = threadIdx.x >> 5;
  const int lane = threadIdx.x & 31;
  const int hi   = lane >> 4;
  const int ln   = lane & 15;
  const int bh   = b * HEADS + h;
  const int q0   = blockIdx.x * 64 + wave * 16;

  const uint32_t ldsK_off = (uint32_t)(uintptr_t)(void*)&ldsK[0];
  const uint32_t ldsV_off = (uint32_t)(uintptr_t)(void*)&ldsV[0];

  // Q strip: two 16x32 A-fragments (d 0..31, 32..63), loaded once.
  const _Float16* qptr = Qh + ((size_t)bh * SEQ + q0 + ln) * HD;
  const v16h qa0 = load_frag_f16(qptr, hi);
  const v16h qa1 = load_frag_f16(qptr + 32, hi);

  float m[8], l[8], alpha[8];
  v8f   o[4] = {};
#pragma unroll
  for (int r = 0; r < 8; ++r) { m[r] = -1.0e30f; l[r] = 0.0f; }

  _Float16* pbuf = &ldsP[wave][0];

  for (int j0 = 0; j0 < SEQ; j0 += 32) {
    // ---- TDM: stage K tile (32 rows x 64) and V tile (64 rows x 32) ----
    if (wave == 0) {
      const uint64_t kaddr =
          (uint64_t)(uintptr_t)(Kh + ((size_t)bh * SEQ + j0) * HD);
      const uint64_t vaddr =
          (uint64_t)(uintptr_t)(Vt + (size_t)bh * HD * SEQ + j0);
      u32x4 kg0 = tdm_group0(ldsK_off, kaddr);
      u32x8 kg1 = tdm_group1(/*tensor_d0=*/HD, /*tensor_d1=*/SEQ,
                             /*tile_d0=*/HD, /*tile_d1=*/32, /*stride0=*/HD);
      u32x4 vg0 = tdm_group0(ldsV_off, vaddr);
      u32x8 vg1 = tdm_group1(/*tensor_d0=*/SEQ, /*tensor_d1=*/HD,
                             /*tile_d0=*/32, /*tile_d1=*/HD, /*stride0=*/SEQ);
      asm volatile("tensor_load_to_lds %0, %1" :: "s"(kg0), "s"(kg1) : "memory");
      asm volatile("tensor_load_to_lds %0, %1" :: "s"(vg0), "s"(vg1) : "memory");
      __builtin_amdgcn_s_wait_tensorcnt(0);
    }
    __syncthreads();

    // ---- scores: two 16x16 j-subtiles, each accumulating two d-chunks ----
    v16h kb00 = load_frag_f16(ldsK + ln * HD, hi);
    v16h kb01 = load_frag_f16(ldsK + ln * HD + 32, hi);
    v16h kb10 = load_frag_f16(ldsK + (16 + ln) * HD, hi);
    v16h kb11 = load_frag_f16(ldsK + (16 + ln) * HD + 32, hi);

    v8f s0 = {};
    s0 = WMMA_F16(qa0, kb00, s0);
    s0 = WMMA_F16(qa1, kb01, s0);
    v8f s1 = {};
    s1 = WMMA_F16(qa0, kb10, s1);
    s1 = WMMA_F16(qa1, kb11, s1);

    // ---- mask + online softmax (each row spans 16 lanes of a half) ----
#pragma unroll
    for (int r = 0; r < 8; ++r) {
      const int  row = q0 + r + hi * 8;
      const int* mp  = mask + ((size_t)b * SEQ + row) * SEQ + j0 + ln;
      float x0 = s0[r] + (float)(1 - mp[0])  * -10000.0f;
      float x1 = s1[r] + (float)(1 - mp[16]) * -10000.0f;

      float mx = fmaxf(x0, x1);
#pragma unroll
      for (int off = 1; off < 16; off <<= 1)
        mx = fmaxf(mx, __shfl_xor(mx, off, 32));
      const float nm = fmaxf(m[r], mx);

      const float p0 = __expf(x0 - nm);
      const float p1 = __expf(x1 - nm);
      float rs = p0 + p1;
#pragma unroll
      for (int off = 1; off < 16; off <<= 1)
        rs += __shfl_xor(rs, off, 32);

      alpha[r] = __expf(m[r] - nm);
      l[r]     = l[r] * alpha[r] + rs;
      m[r]     = nm;

      // C-layout -> LDS (row-major 16x32 P tile, wave-private)
      const int prow = r + hi * 8;
      pbuf[prow * 32 + ln]      = (_Float16)p0;
      pbuf[prow * 32 + 16 + ln] = (_Float16)p1;
    }

    // rescale running output accumulators
#pragma unroll
    for (int tt = 0; tt < 4; ++tt)
#pragma unroll
      for (int r = 0; r < 8; ++r) o[tt][r] *= alpha[r];

    // per-wave LDS ops are in-order; wait for P stores before A-layout reload
    asm volatile("s_wait_dscnt 0" ::: "memory");
    const v16h pa = load_frag_f16(pbuf + ln * 32, hi);

    // ---- O += P x V : 4 d-tiles of 16, K-dim = 32 keys (from LDS) ----
#pragma unroll
    for (int tt = 0; tt < 4; ++tt) {
      v16h vfrag = load_frag_f16(ldsV + (tt * 16 + ln) * 32, hi);
      o[tt] = WMMA_F16(pa, vfrag, o[tt]);
    }

    __syncthreads();   // all waves done with ldsK/ldsV before next TDM write
  }

  // ---- finalize: divide by l, write f16 rows [b][s][h*64+d] ----
#pragma unroll
  for (int r = 0; r < 8; ++r) {
    const float inv = 1.0f / l[r];
    const int   row = q0 + r + hi * 8;
    _Float16* op = O16 + ((size_t)b * SEQ + row) * DIM + h * HD;
#pragma unroll
    for (int tt = 0; tt < 4; ++tt)
      op[tt * 16 + ln] = (_Float16)(o[tt][r] * inv);
  }
}

// ---------------------------------------------------------------------------
// Kernel 3: output projection  Y = O16 @ Wo + bo  (fp32 out), 16x32 per wave
// ---------------------------------------------------------------------------
__global__ __launch_bounds__(256) void out_proj_kernel(
    const _Float16* __restrict__ A16, const _Float16* __restrict__ Wto,
    const float* __restrict__ bo, float* __restrict__ Y)
{
  const int wave = threadIdx.x >> 5;
  const int lane = threadIdx.x & 31;
  const int hi   = lane >> 4;
  const int ln   = lane & 15;

  const int t  = blockIdx.x * 8 + wave;          // 12288 tiles total
  const int tm = (t / (DIM / 32)) * 16;
  const int tn = (t % (DIM / 32)) * 32;

  const _Float16* arow = A16 + (size_t)(tm + ln) * DIM;
  const _Float16* w0p  = Wto + (size_t)(tn + ln) * DIM;
  const _Float16* w1p  = Wto + (size_t)(tn + 16 + ln) * DIM;

  v8f acc0 = {}, acc1 = {};
#pragma unroll 4
  for (int k0 = 0; k0 < DIM; k0 += 32) {
    v16h a  = load_frag_f16(arow + k0, hi);
    v16h b0 = load_frag_f16(w0p + k0, hi);
    v16h b1 = load_frag_f16(w1p + k0, hi);
    acc0 = WMMA_F16(a, b0, acc0);
    acc1 = WMMA_F16(a, b1, acc1);
  }

#pragma unroll
  for (int half = 0; half < 2; ++half) {
    const int   col = tn + half * 16 + ln;
    const float bb  = bo[col];
    const v8f&  acc = half ? acc1 : acc0;
#pragma unroll
    for (int r = 0; r < 8; ++r) {
      const int row = tm + r + hi * 8;
      Y[(size_t)row * DIM + col] = acc[r] + bb;
    }
  }
}

// ---------------------------------------------------------------------------
extern "C" void kernel_launch(void* const* d_in, const int* in_sizes, int n_in,
                              void* d_out, int out_size, void* d_ws, size_t ws_size,
                              hipStream_t stream) {
  (void)in_sizes; (void)n_in; (void)out_size; (void)ws_size;

  const float* from_t = (const float*)d_in[0];
  const float* to_t   = (const float*)d_in[1];
  const int*   mask   = (const int*)d_in[2];
  const float* Wq = (const float*)d_in[3];
  const float* bq = (const float*)d_in[4];
  const float* Wk = (const float*)d_in[5];
  const float* bk = (const float*)d_in[6];
  const float* Wv = (const float*)d_in[7];
  const float* bv = (const float*)d_in[8];
  const float* Wo = (const float*)d_in[9];
  const float* bo = (const float*)d_in[10];
  float* out = (float*)d_out;

  // workspace layout (halves):
  //  Qh | Kh | Vt | O16 | Xf | Xt   each NQ = B*H*S*64 = 6,291,456
  //  Wtq | Wtk | Wtv | Wto          each 768*768 = 589,824
  const size_t NQ = (size_t)NB * HEADS * SEQ * HD;
  const size_t NW = (size_t)DIM * DIM;
  _Float16* Qh  = (_Float16*)d_ws;
  _Float16* Kh  = Qh + NQ;
  _Float16* Vt  = Kh + NQ;
  _Float16* O16 = Vt + NQ;
  _Float16* Xf  = O16 + NQ;
  _Float16* Xt  = Xf + NQ;
  _Float16* Wtq = Xt + NQ;
  _Float16* Wtk = Wtq + NW;
  _Float16* Wtv = Wtk + NW;
  _Float16* Wto = Wtv + NW;

  // 0) one-shot packs: activations f16, weights f16-transposed
  cvt_f32_f16_kernel<<<dim3((int)(NQ / 256)), 256, 0, stream>>>(from_t, Xf, (int)NQ);
  cvt_f32_f16_kernel<<<dim3((int)(NQ / 256)), 256, 0, stream>>>(to_t,   Xt, (int)NQ);
  transpose_cvt_kernel<<<dim3((int)(NW / 256)), 256, 0, stream>>>(Wq, Wtq);
  transpose_cvt_kernel<<<dim3((int)(NW / 256)), 256, 0, stream>>>(Wk, Wtk);
  transpose_cvt_kernel<<<dim3((int)(NW / 256)), 256, 0, stream>>>(Wv, Wtv);
  transpose_cvt_kernel<<<dim3((int)(NW / 256)), 256, 0, stream>>>(Wo, Wto);

  // 1) QKV projections: (SEQ/16)*(DIM/32)=3072 tiles per (b,which), 8 waves/blk
  qkv_proj_kernel<<<dim3(384, NB, 3), 256, 0, stream>>>(
      Xf, Xt, Wtq, bq, Wtk, bk, Wtv, bv, Qh, Kh, Vt);

  // 2) flash attention: 64 Q rows per block (4 waves x 16 rows)
  flash_attn_kernel<<<dim3(SEQ / 64, HEADS, NB), 128, 0, stream>>>(
      Qh, Kh, Vt, mask, O16);

  // 3) output projection: (B*SEQ/16)*(DIM/32)=12288 tiles, 8 waves/blk
  out_proj_kernel<<<dim3(1536), 256, 0, stream>>>(O16, Wto, bo, out);
}
